// refinedet_multibox_loss_35871566856709
// MI455X (gfx1250) — compile-verified, matrix operand order checked by reading, not run on hardware
//
#include <hip/hip_runtime.h>
#include <cstdint>

// Problem constants (from the reference)
static constexpr int kNB = 64;      // batch
static constexpr int kP  = 32768;   // priors
static constexpr int kO  = 50;      // objects per image
static constexpr float kThresh = 0.5f;
static constexpr int   kNegPos = 3;
static constexpr float kVar0 = 0.1f;
static constexpr float kVar1 = 0.2f;

typedef __attribute__((ext_vector_type(16))) _Float16 v16h;
typedef __attribute__((ext_vector_type(8)))  float    v8f;

// ---------------------------------------------------------------- zero scratch
__global__ void zero_ws(uint32_t* __restrict__ p, int n) {
  for (int i = blockIdx.x * blockDim.x + threadIdx.x; i < n;
       i += gridDim.x * blockDim.x)
    p[i] = 0u;
}

// ---------------------------------------------------------------- matching
// One prior per thread; truths staged in LDS via async-to-LDS.
// Per-prior best truth -> packed (iou_bits & ~63) | truth_idx.
// Per-truth best prior -> u64 atomicMax key (iou_bits<<32 | prior_idx).
__global__ __launch_bounds__(256)
void match_kernel(const float4* __restrict__ priors4,
                  const float*  __restrict__ targets,
                  uint32_t* __restrict__ bt_packed,
                  unsigned long long* __restrict__ bestprior)
{
  __shared__ float traw[kO * 5];
  __shared__ float tarea[kO];
  __shared__ unsigned long long lbest[kO];

  const int b   = blockIdx.y;
  const int tid = threadIdx.x;
  const float* gsrc = targets + (size_t)b * kO * 5;

  if (tid < kO * 5) {
#if defined(__HIP_DEVICE_COMPILE__) && __has_builtin(__builtin_amdgcn_global_load_async_to_lds_b32)
    __builtin_amdgcn_global_load_async_to_lds_b32(
        (__attribute__((address_space(1))) int*)(gsrc + tid),
        (__attribute__((address_space(3))) int*)(&traw[tid]), 0, 0);
#else
    traw[tid] = gsrc[tid];
#endif
  }
#if defined(__HIP_DEVICE_COMPILE__) && __has_builtin(__builtin_amdgcn_global_load_async_to_lds_b32)
#if __has_builtin(__builtin_amdgcn_s_wait_asynccnt)
  __builtin_amdgcn_s_wait_asynccnt(0);
#else
  asm volatile("s_wait_asynccnt 0x0" ::: "memory");
#endif
#endif
  if (tid < kO) lbest[tid] = 0ull;
  __syncthreads();
  if (tid < kO) {
    float x1 = traw[tid*5+0], y1 = traw[tid*5+1];
    float x2 = traw[tid*5+2], y2 = traw[tid*5+3];
    tarea[tid] = (x2 - x1) * (y2 - y1);
  }
  __syncthreads();

  const int j = blockIdx.x * 256 + tid;
  float4 pr = priors4[j];
  const float px1 = pr.x - pr.z * 0.5f, py1 = pr.y - pr.w * 0.5f;
  const float px2 = pr.x + pr.z * 0.5f, py2 = pr.y + pr.w * 0.5f;
  const float parea = pr.z * pr.w;

  float best_o = -1.0f;
  int   best_t = 0;
  for (int t = 0; t < kO; ++t) {
    float tx1 = traw[t*5+0], ty1 = traw[t*5+1];
    float tx2 = traw[t*5+2], ty2 = traw[t*5+3];
    float ix = fminf(px2, tx2) - fmaxf(px1, tx1);
    float iy = fminf(py2, ty2) - fmaxf(py1, ty1);
    ix = fmaxf(ix, 0.0f); iy = fmaxf(iy, 0.0f);
    float inter = ix * iy;
    float ov = inter / (parea + tarea[t] - inter);
    if (ov > best_o) { best_o = ov; best_t = t; }
    unsigned long long key =
        ((unsigned long long)__float_as_uint(ov) << 32) | (unsigned)j;
    atomicMax(&lbest[t], key);   // ds_max_u64 — stays in LDS
  }
  best_o = fmaxf(best_o, 0.0f);
  bt_packed[(size_t)b * kP + j] =
      (__float_as_uint(best_o) & 0xFFFFFFC0u) | (unsigned)best_t;

  __syncthreads();
  if (tid < kO) atomicMax(&bestprior[b * 64 + tid], lbest[tid]);
}

// ---------------------------------------------------------------- override
// best_truth_ovl[best_prior_idx[t]] = 2.0; idx = t (ascending t => last wins)
__global__ void override_kernel(const unsigned long long* __restrict__ bestprior,
                                uint32_t* __restrict__ bt_packed)
{
  const int b = blockIdx.x;
  if (threadIdx.x != 0) return;
  const unsigned two = __float_as_uint(2.0f);   // low 6 bits are zero
  for (int t = 0; t < kO; ++t) {
    unsigned j = (unsigned)(bestprior[b * 64 + t] & 0xFFFFFFFFull);
    bt_packed[(size_t)b * kP + j] = two | (unsigned)t;
  }
}

// ---------------------------------------------------------------- main loss pass
// Smooth-L1 + conf logsumexp; per-wave totals reduced through the WMMA pipe:
// D = A(16x32 f16 of per-prior terms) x ones(32x16) -> f32 row sums.
// Lane L holds rows 0-7 of one column, lane L+16 rows 8-15 of the same
// column, so one shfl_xor(16) finishes the grand total (no /16 needed).
__global__ __launch_bounds__(256)
void loss_kernel(const float4* __restrict__ arm_loc4,
                 const float2* __restrict__ arm_conf2,
                 const float4* __restrict__ priors4,
                 const float*  __restrict__ targets,
                 const uint32_t* __restrict__ bt_packed,
                 float* __restrict__ lc_arr,
                 uint32_t* __restrict__ hist,
                 uint32_t* __restrict__ num_pos,
                 float* __restrict__ loss_acc)   // [0]=loss_l  [1]=loss_c
{
  const int b    = blockIdx.y;
  const int tid  = threadIdx.x;
  const int base = blockIdx.x * 4096;

  v16h a_l, a_c, onesv;
  for (int i = 0; i < 16; ++i) onesv[i] = (_Float16)1.0f;

  unsigned poscnt = 0;
  for (int i = 0; i < 16; ++i) {
    const int j = base + i * 256 + tid;
    const size_t gj = (size_t)b * kP + j;
    if (i < 15) {   // stream-ahead: global_prefetch_b8
      __builtin_prefetch(&arm_loc4[gj + 256], 0, 1);
      __builtin_prefetch(&arm_conf2[gj + 256], 0, 1);
    }
    const uint32_t key = bt_packed[gj];
    const int   t   = (int)(key & 63u);
    const float ovl = __uint_as_float(key & 0xFFFFFFC0u);
    const float* tr = targets + ((size_t)b * kO + t) * 5;
    const int label  = (tr[4] >= 0.0f) ? 1 : 0;
    const int conf_t = (ovl < kThresh) ? 0 : label;
    const bool pos = conf_t > 0;

    float sl = 0.0f;
    if (pos) {
      float4 prr = priors4[j];
      float mx1 = tr[0], my1 = tr[1], mx2 = tr[2], my2 = tr[3];
      float gcx = ((mx1 + mx2) * 0.5f - prr.x) / (kVar0 * prr.z);
      float gcy = ((my1 + my2) * 0.5f - prr.y) / (kVar0 * prr.w);
      float gw  = __logf((mx2 - mx1) / prr.z) / kVar1;
      float gh  = __logf((my2 - my1) / prr.w) / kVar1;
      float4 L = arm_loc4[gj];
      float d0 = L.x - gcx, d1 = L.y - gcy, d2 = L.z - gw, d3 = L.w - gh;
      float a0 = fabsf(d0), a1 = fabsf(d1), a2 = fabsf(d2), a3 = fabsf(d3);
      sl = (a0 < 1.f ? 0.5f * d0 * d0 : a0 - 0.5f)
         + (a1 < 1.f ? 0.5f * d1 * d1 : a1 - 0.5f)
         + (a2 < 1.f ? 0.5f * d2 * d2 : a2 - 0.5f)
         + (a3 < 1.f ? 0.5f * d3 * d3 : a3 - 0.5f);
    }

    float2 Cc = arm_conf2[gj];
    float m   = fmaxf(Cc.x, Cc.y);
    float lse = m + __logf(__expf(Cc.x - m) + __expf(Cc.y - m));
    float picked = conf_t ? Cc.y : Cc.x;
    float lc = lse - picked;
    float lc_mine = pos ? 0.0f : lc;
    lc_arr[gj] = lc_mine;
    atomicAdd(&hist[b * 4096 + (int)(__float_as_uint(lc_mine) >> 20)], 1u);

    a_l[i] = (_Float16)sl;
    a_c[i] = (_Float16)(pos ? lc : 0.0f);
    poscnt += (unsigned)__popcll(__ballot(pos));
  }

  v8f c_l = {}; v8f c_c = {};
  c_l = __builtin_amdgcn_wmma_f32_16x16x32_f16(false, a_l, false, onesv,
                                               (short)0, c_l, false, false);
  c_c = __builtin_amdgcn_wmma_f32_16x16x32_f16(false, a_c, false, onesv,
                                               (short)0, c_c, false, false);
  float s_l = 0.f, s_c = 0.f;
  for (int v = 0; v < 8; ++v) { s_l += c_l[v]; s_c += c_c[v]; }
  // rows 0-7 (lane L) + rows 8-15 (lane L+16) of the same column => total
  s_l += __shfl_xor(s_l, 16, 32);
  s_c += __shfl_xor(s_c, 16, 32);
  if ((tid & 31) == 0) {
    atomicAdd(&loss_acc[0], s_l);
    atomicAdd(&loss_acc[1], s_c);
    atomicAdd(&num_pos[b], poscnt);
  }
}

// ---------------------------------------------------------------- find top-K cut
__global__ void cut_kernel(const uint32_t* __restrict__ hist,
                           const uint32_t* __restrict__ num_pos,
                           int* __restrict__ cut_bin,
                           uint32_t* __restrict__ remv)
{
  const int b = blockIdx.x * blockDim.x + threadIdx.x;
  if (b >= kNB) return;
  long long K = (long long)kNegPos * (long long)num_pos[b];
  if (K > kP - 1) K = kP - 1;
  const uint32_t* h = hist + b * 4096;
  long long acc = 0, rem = 0;
  int cut = 0; bool found = false;
  for (int bin = 4095; bin >= 1; --bin) {
    uint32_t c = h[bin];
    if (acc + (long long)c >= K) { cut = bin; rem = K - acc; found = true; break; }
    acc += c;
  }
  if (!found) { cut = 0; rem = K - acc; }
  if (rem < 0) rem = 0;
  cut_bin[b] = cut;
  remv[b]    = (uint32_t)rem;
}

// ---------------------------------------------------------------- negative sum
__global__ __launch_bounds__(256)
void negsum_kernel(const float* __restrict__ lc_arr,
                   const int* __restrict__ cut_bin,
                   const uint32_t* __restrict__ remv,
                   uint32_t* __restrict__ taken,
                   float* __restrict__ loss_acc)
{
  const int b = blockIdx.y;
  const int j = blockIdx.x * 256 + threadIdx.x;
  const size_t gj = (size_t)b * kP + j;
  const float lc = lc_arr[gj];
  const int bin = (int)(__float_as_uint(lc) >> 20);
  const int cut = cut_bin[b];
  float v = 0.0f;
  if (bin > cut) {
    v = lc;
  } else if (bin == cut) {
    uint32_t slot = atomicAdd(&taken[b], 1u);
    if (slot < remv[b]) v = lc;
  }
  for (int off = 16; off > 0; off >>= 1) v += __shfl_xor(v, off, 32);
  if ((threadIdx.x & 31) == 0 && v != 0.0f) atomicAdd(&loss_acc[1], v);
}

// ---------------------------------------------------------------- finalize
__global__ void final_kernel(const uint32_t* __restrict__ num_pos,
                             const float* __restrict__ loss_acc,
                             float* __restrict__ out)
{
  if (threadIdx.x == 0 && blockIdx.x == 0) {
    unsigned n = 0;
    for (int b = 0; b < kNB; ++b) n += num_pos[b];
    const float N = (float)n;
    out[0] = loss_acc[0] / N;
    out[1] = loss_acc[1] / N;
  }
}

// ---------------------------------------------------------------- launch
extern "C" void kernel_launch(void* const* d_in, const int* in_sizes, int n_in,
                              void* d_out, int out_size, void* d_ws, size_t ws_size,
                              hipStream_t stream)
{
  (void)in_sizes; (void)n_in; (void)out_size; (void)ws_size;
  const float* arm_loc  = (const float*)d_in[0];   // (B,P,4)
  const float* arm_conf = (const float*)d_in[1];   // (B,P,2)
  const float* priors   = (const float*)d_in[4];   // (P,4)
  const float* targets  = (const float*)d_in[5];   // (B,O,5)

  char* ws = (char*)d_ws;
  uint32_t* bt_packed = (uint32_t*)ws;                          // B*P*4
  float*    lc_arr    = (float*)(ws + (size_t)kNB * kP * 4);    // B*P*4
  char*     zbase     = ws + (size_t)kNB * kP * 8;              // zeroed region:
  uint32_t* hist      = (uint32_t*)zbase;                       //  B*4096*4
  unsigned long long* bestprior =
      (unsigned long long*)(zbase + (size_t)kNB * 4096 * 4);    //  B*64*8
  uint32_t* num_pos = (uint32_t*)((char*)bestprior + (size_t)kNB * 64 * 8);
  int*      cut_bin = (int*)((char*)num_pos + kNB * 4);
  uint32_t* remv    = (uint32_t*)((char*)cut_bin + kNB * 4);
  uint32_t* taken   = (uint32_t*)((char*)remv + kNB * 4);
  float*    loss_acc= (float*)((char*)taken + kNB * 4);

  const int zero_words =
      (kNB * 4096 * 4 + kNB * 64 * 8 + 4 * kNB * 4 + 8) / 4;

  zero_ws<<<256, 256, 0, stream>>>((uint32_t*)zbase, zero_words);
  match_kernel<<<dim3(kP / 256, kNB), 256, 0, stream>>>(
      (const float4*)priors, targets, bt_packed, bestprior);
  override_kernel<<<kNB, 32, 0, stream>>>(bestprior, bt_packed);
  loss_kernel<<<dim3(kP / 4096, kNB), 256, 0, stream>>>(
      (const float4*)arm_loc, (const float2*)arm_conf,
      (const float4*)priors, targets, bt_packed,
      lc_arr, hist, num_pos, loss_acc);
  cut_kernel<<<1, 64, 0, stream>>>(hist, num_pos, cut_bin, remv);
  negsum_kernel<<<dim3(kP / 256, kNB), 256, 0, stream>>>(
      lc_arr, cut_bin, remv, taken, loss_acc);
  final_kernel<<<1, 1, 0, stream>>>(num_pos, loss_acc, (float*)d_out);
}
